// FlashAttention_71725953843358
// MI455X (gfx1250) — compile-verified, matrix operand order checked by reading, not run on hardware
//
#include <hip/hip_runtime.h>

// MI455X / gfx1250 fused attention with materialized attention weights.
// out  = softmax(Q K^T * scale) V        [B,H,S,D]
// attn = softmax(Q K^T * scale)          [B,H,S,S]  (streamed to HBM once)
//
// Phase 1: scores via v_wmma_f32_16x16x32_bf16 into a 32x2048 LDS strip.
// Phase 2: row softmax in LDS, normalized P written once to HBM (float4).
// Phase 3: O = P*V with WMMA; V chunks staged into LDS by the Tensor Data
//          Mover (tensor_load_to_lds, TENSORcnt) with double buffering.

typedef __attribute__((ext_vector_type(16))) __bf16       v16bf;
typedef __attribute__((ext_vector_type(8)))  float        v8f;
typedef __attribute__((ext_vector_type(4)))  unsigned int u32x4;
typedef __attribute__((ext_vector_type(8)))  int          i32x8;
typedef __attribute__((ext_vector_type(4)))  int          i32x4;

#define B_    2
#define H_    16
#define S_    2048
#define D_    64
#define QB    32            // query rows per workgroup
#define NSEG  8             // threads per row in softmax phase
#define SCALE 0.125f        // 1/sqrt(64)
#define VCH   32            // V rows per TDM chunk
#define VBUFF (VCH * D_)    // floats per V staging buffer (2048)

// ---- Tensor Data Mover availability / arity detection -----------------------
#if __has_builtin(__builtin_amdgcn_tensor_load_to_lds)
#  define HAS_TDM 1
#  if __has_include(<hip/amd_detail/amd_gfx1250_TDM.h>)
     // amdgpu-toolchain (clang-23 / therock-10.0): 6-arg form
#    define TDM_LOAD(g0, g1, g2, g3, cpol) \
       __builtin_amdgcn_tensor_load_to_lds((g0), (g1), (g2), (g3), \
                                           (i32x8){0,0,0,0,0,0,0,0}, (cpol))
#  else
     // ROCm 7.2 (clang-22): 5-arg form
#    define TDM_LOAD(g0, g1, g2, g3, cpol) \
       __builtin_amdgcn_tensor_load_to_lds((g0), (g1), (g2), (g3), (cpol))
#  endif
#  if __has_builtin(__builtin_amdgcn_s_wait_tensorcnt)
#    define WAIT_TENSOR(n) __builtin_amdgcn_s_wait_tensorcnt((short)(n))
#  else
#    define WAIT_TENSOR(n) asm volatile("s_wait_tensorcnt %0" ::"i"(n))
#  endif
#endif

__global__ void __launch_bounds__(256, 1)
fa_fused_kernel(const float* __restrict__ q,
                const float* __restrict__ k,
                const float* __restrict__ v,
                float* __restrict__ out,
                float* __restrict__ attn) {
    extern __shared__ float lds[];
    float* sc   = lds;                  // QB * S_ floats = 256 KB score strip
    float* redA = lds + QB * S_;        // 256 floats
    float* redB = redA + 256;           // 256 floats
    float* vbuf = redB + 256;           // 2 * VBUFF floats = 16 KB V staging

    const int blk  = blockIdx.x;
    const int bh   = blk / (S_ / QB);
    const int q0   = (blk % (S_ / QB)) * QB;
    const int tid  = threadIdx.x;
    const int lane = tid & 31;
    const int w    = tid >> 5;          // wave id 0..7
    const int m    = lane & 15;         // row (A) / col (B) index within tile
    const int hi   = lane >> 4;         // 0: K-low half, 1: K-high half
    const size_t qkv_base = (size_t)bh * S_ * D_;

    // ---------------- Phase 1: S = scale * Q K^T  (bf16 WMMA) ----------------
    {
        const int qt = w & 1;           // which 16-row qtile of the 32-row block
        const float* qrow = q + qkv_base + (size_t)(q0 + qt * 16 + m) * D_;
        v16bf a0, a1;
        #pragma unroll
        for (int e = 0; e < 16; ++e) {
            const int kk = ((e < 8) ? e : e + 8) + hi * 8;   // 16-bit A striping
            a0[e] = (__bf16)qrow[kk];
            a1[e] = (__bf16)qrow[32 + kk];
        }
        for (int kt = (w >> 1); kt < S_ / 16; kt += 4) {
            const float* krow = k + qkv_base + (size_t)(kt * 16 + m) * D_;
            __builtin_prefetch(krow + (size_t)4 * 16 * D_, 0, 1);
            v16bf b0, b1;
            #pragma unroll
            for (int e = 0; e < 16; ++e) {
                const int kk = ((e < 8) ? e : e + 8) + hi * 8;
                b0[e] = (__bf16)krow[kk];
                b1[e] = (__bf16)krow[32 + kk];
            }
            v8f c = {};
            c = __builtin_amdgcn_wmma_f32_16x16x32_bf16(false, a0, false, b0,
                                                        (short)0, c, false, false);
            c = __builtin_amdgcn_wmma_f32_16x16x32_bf16(false, a1, false, b1,
                                                        (short)0, c, false, false);
            #pragma unroll
            for (int r = 0; r < 8; ++r)
                sc[(size_t)(qt * 16 + r + hi * 8) * S_ + kt * 16 + m] = c[r] * SCALE;
        }
    }
    __syncthreads();

    // ---------------- Phase 2: row softmax in LDS, stream P to HBM ----------------
    {
        const int r   = tid >> 3;                 // row 0..31
        const int seg = tid & (NSEG - 1);         // 8 segments of 256 cols
        const int c0  = seg * (S_ / NSEG);
        float* row = sc + (size_t)r * S_;

        float mx = -3.402823466e38f;
        for (int c = c0; c < c0 + S_ / NSEG; c += 4) {
            const float4 p = *(const float4*)(row + c);
            mx = fmaxf(mx, fmaxf(fmaxf(p.x, p.y), fmaxf(p.z, p.w)));
        }
        redA[tid] = mx;
        __syncthreads();
        #pragma unroll
        for (int j = 0; j < NSEG; ++j) mx = fmaxf(mx, redA[(r << 3) + j]);

        float sum = 0.f;
        for (int c = c0; c < c0 + S_ / NSEG; c += 4) {
            float4 p = *(const float4*)(row + c);
            p.x = __expf(p.x - mx); p.y = __expf(p.y - mx);
            p.z = __expf(p.z - mx); p.w = __expf(p.w - mx);
            *(float4*)(row + c) = p;
            sum += p.x + p.y + p.z + p.w;
        }
        redB[tid] = sum;
        __syncthreads();
        sum = 0.f;
        #pragma unroll
        for (int j = 0; j < NSEG; ++j) sum += redB[(r << 3) + j];
        const float inv = 1.0f / sum;

        float* gattn = attn + (size_t)bh * S_ * S_ + (size_t)(q0 + r) * S_;
        for (int c = c0; c < c0 + S_ / NSEG; c += 4) {
            float4 p = *(const float4*)(row + c);
            p.x *= inv; p.y *= inv; p.z *= inv; p.w *= inv;
            *(float4*)(row + c)   = p;    // normalized P for phase 3
            *(float4*)(gattn + c) = p;    // the one mandatory 537MB write
        }
    }
    __syncthreads();

    // ---------------- Phase 3: O = P V  (bf16 WMMA) ----------------
    {
        const int qt = w >> 2;          // 0..1   (16-row M tile)
        const int nt = w & 3;           // 0..3   (16-col N tile of D=64)
        v8f acc = {};

#ifdef HAS_TDM
        // ---- TDM path: DMA each 32x64 V chunk into LDS, double-buffered ----
        const unsigned      lds_vb = (unsigned)(uintptr_t)(const void*)vbuf;
        const unsigned long long vga =
            (unsigned long long)(uintptr_t)(const void*)(v + qkv_base);

        auto tdm_issue = [&](int kc, int buf) {
            const unsigned long long ga =
                vga + (unsigned long long)kc * VCH * D_ * sizeof(float);
            u32x4 g0;
            g0.x = 1u;                                     // count=1, user mode
            g0.y = lds_vb + (unsigned)buf * (VBUFF * 4);   // lds_addr (bytes)
            g0.z = (unsigned)ga;                           // global_addr lo32
            g0.w = ((unsigned)(ga >> 32) & 0x01FFFFFFu)    // global_addr hi
                   | 0x80000000u;                          // type=2 ("image")
            i32x8 g1;
            g1[0] = 0x00020000;                            // data_size=4B
            g1[1] = (int)(D_ << 16);                       // tensor_dim0 = 64
            g1[2] = (int)((unsigned)S_ << 16);             // tensor_dim1 = 2048
            g1[3] = (int)(D_ << 16);                       // tile_dim0 = 64
            g1[4] = VCH;                                   // tile_dim1 = 32
            g1[5] = D_;                                    // tensor_dim0_stride
            g1[6] = 0;
            g1[7] = 0;
            const i32x4 z = {0, 0, 0, 0};
            TDM_LOAD(g0, g1, z, z, 0);
        };

        if (w == 0) tdm_issue(0, 0);
        for (int kc = 0; kc < S_ / VCH; ++kc) {
            if (w == 0) {
                if (kc + 1 < S_ / VCH) { tdm_issue(kc + 1, (kc + 1) & 1); WAIT_TENSOR(1); }
                else                   { WAIT_TENSOR(0); }
            }
            __syncthreads();                       // chunk kc visible in LDS
            const float* vb   = vbuf + (size_t)(kc & 1) * VBUFF;
            const float* prow = sc + (size_t)(qt * 16 + m) * S_ + kc * VCH;
            v16bf a, b;
            #pragma unroll
            for (int e = 0; e < 16; ++e) {
                const int kk = ((e < 8) ? e : e + 8) + hi * 8;
                a[e] = (__bf16)prow[kk];
                b[e] = (__bf16)vb[kk * D_ + nt * 16 + m];
            }
            acc = __builtin_amdgcn_wmma_f32_16x16x32_bf16(false, a, false, b,
                                                          (short)0, acc, false, false);
            __syncthreads();                       // done reading buf before reuse
        }
#else
        // ---- Fallback: read V directly from global ----
        for (int kc = 0; kc < S_ / VCH; ++kc) {
            const float* prow  = sc + (size_t)(qt * 16 + m) * S_ + kc * VCH;
            const float* vbase = v + qkv_base + (size_t)(kc * VCH) * D_ + nt * 16 + m;
            v16bf a, b;
            #pragma unroll
            for (int e = 0; e < 16; ++e) {
                const int kk = ((e < 8) ? e : e + 8) + hi * 8;
                a[e] = (__bf16)prow[kk];
                b[e] = (__bf16)vbase[(size_t)kk * D_];
            }
            acc = __builtin_amdgcn_wmma_f32_16x16x32_bf16(false, a, false, b,
                                                          (short)0, acc, false, false);
        }
#endif
        float* orow = out + qkv_base + (size_t)(q0 + qt * 16 + hi * 8) * D_ + nt * 16 + m;
        #pragma unroll
        for (int r = 0; r < 8; ++r)
            orow[(size_t)r * D_] = acc[r];
    }
}

extern "C" void kernel_launch(void* const* d_in, const int* in_sizes, int n_in,
                              void* d_out, int out_size, void* d_ws, size_t ws_size,
                              hipStream_t stream) {
    const float* q = (const float*)d_in[0];
    const float* k = (const float*)d_in[1];
    const float* v = (const float*)d_in[2];
    float* out  = (float*)d_out;
    float* attn = out + (size_t)B_ * H_ * S_ * D_;   // outputs concatenated flat

    const dim3 grid(B_ * H_ * (S_ / QB));            // 2048 workgroups
    const dim3 block(256);                           // 8 wave32
    const size_t shmem =
        ((size_t)QB * S_ + 512 + 2 * VBUFF) * sizeof(float); // ~280.5 KB LDS

    hipLaunchKernelGGL(fa_fused_kernel, grid, block, shmem, stream,
                       q, k, v, out, attn);
}
